// SelfAttention_6305011990639
// MI455X (gfx1250) — matrix operand order repaired and checked by reading, not compiled, not run on hardware
//
#include <hip/hip_runtime.h>
#include <hip/hip_bf16.h>

#define EMBED 1024
#define HEADS 16
#define HDIM  64
#define NB    4
#define LSEQ  2048

typedef __bf16 bf16x16 __attribute__((ext_vector_type(16)));
typedef float  f32x8   __attribute__((ext_vector_type(8)));
typedef float  f32x4   __attribute__((ext_vector_type(4)));
typedef unsigned short u16x8 __attribute__((ext_vector_type(8)));

union FragU { bf16x16 bf; u16x8 u[2]; };
union AccU  { f32x8 v; float f[8]; };

__device__ __forceinline__ unsigned short f2bf(float x) {
  unsigned u = __float_as_uint(x);
  unsigned r = u + 0x7FFFu + ((u >> 16) & 1u);   // round-to-nearest-even
  return (unsigned short)(r >> 16);
}

__device__ __forceinline__ f32x8 wmma_bf(bf16x16 a, bf16x16 b, f32x8 c) {
  return __builtin_amdgcn_wmma_f32_16x16x32_bf16(false, a, false, b, (short)0, c,
                                                 false, false);
}

// A-matrix fragment (16x32, 16-bit): lane = row m, elements cover two 8-wide
// contiguous K runs at koff+8*half and koff+16+8*half (ISA 7.12.2 table).
__device__ __forceinline__ bf16x16 fragA_bf16(const unsigned short* base, int stride,
                                              int koff, int lane) {
  int m = lane & 15, hf = (lane >> 4) & 1;
  const unsigned short* p = base + (size_t)m * stride + koff + 8 * hf;
  FragU f;
  f.u[0] = *(const u16x8*)(p);
  f.u[1] = *(const u16x8*)(p + 16);
  return f.bf;
}

// B-matrix fragment (32x16, 16-bit): lane = column n, elements cover two 8-wide
// contiguous K runs at koff+8*half and koff+16+8*half (same order as A).
__device__ __forceinline__ bf16x16 fragB_bf16(const unsigned short* base, int stride,
                                              int koff, int lane) {
  int n = lane & 15, hf = (lane >> 4) & 1;
  const unsigned short* p = base + (size_t)n * stride + koff + 8 * hf;
  FragU f;
  f.u[0] = *(const u16x8*)(p);
  f.u[1] = *(const u16x8*)(p + 16);
  return f.bf;
}

__device__ __forceinline__ u16x8 cvt8(f32x4 a, f32x4 b) {
  u16x8 r;
  r[0] = f2bf(a[0]); r[1] = f2bf(a[1]); r[2] = f2bf(a[2]); r[3] = f2bf(a[3]);
  r[4] = f2bf(b[0]); r[5] = f2bf(b[1]); r[6] = f2bf(b[2]); r[7] = f2bf(b[3]);
  return r;
}

__device__ __forceinline__ bf16x16 fragA_f32(const float* base, int stride,
                                             int koff, int lane) {
  int m = lane & 15, hf = (lane >> 4) & 1;
  const float* p = base + (size_t)m * stride + koff + 8 * hf;
  FragU f;
  f.u[0] = cvt8(*(const f32x4*)(p),      *(const f32x4*)(p + 4));
  f.u[1] = cvt8(*(const f32x4*)(p + 16), *(const f32x4*)(p + 20));
  return f.bf;
}

__device__ __forceinline__ bf16x16 fragB_f32(const float* base, int stride,
                                             int koff, int lane) {
  int n = lane & 15, hf = (lane >> 4) & 1;
  const float* p = base + (size_t)n * stride + koff + 8 * hf;
  FragU f;
  f.u[0] = cvt8(*(const f32x4*)(p),     *(const f32x4*)(p + 4));
  f.u[1] = cvt8(*(const f32x4*)(p + 16), *(const f32x4*)(p + 20));
  return f.bf;
}

// ---------------------------------------------------------------------------
// Tensor Data Mover: 2-D tile (2-byte elements) global -> LDS, per ISA 8.3/8.4.
// This toolchain exposes the 6-arg builtin:
//   (u32x4 g0, i32x8 g1, i32x4 g2, i32x4 g3, i32x8 extra, i32 cpol)
#if __has_builtin(__builtin_amdgcn_tensor_load_to_lds)
#define HAVE_TDM 1
typedef unsigned int u32x4 __attribute__((ext_vector_type(4)));
typedef int i32x8 __attribute__((ext_vector_type(8)));
typedef int i32x4 __attribute__((ext_vector_type(4)));

__device__ __forceinline__ void tdm_load_2d(const unsigned short* lds_dst,
                                            const unsigned short* gsrc,
                                            unsigned tile_w, unsigned tile_h,
                                            unsigned stride_elems,
                                            unsigned tensor_rows) {
  unsigned long long ga = (unsigned long long)(const void*)gsrc;
  unsigned lds_off = (unsigned)(unsigned long long)(const void*)lds_dst;
  u32x4 g0;
  g0[0] = 1u;                                            // count=1, no gather
  g0[1] = lds_off;                                       // lds_addr (bytes)
  g0[2] = (unsigned)ga;                                  // global_addr[31:0]
  g0[3] = (unsigned)((ga >> 32) & 0x01FFFFFFu) | (2u << 30);  // [56:32] | type=2
  i32x8 g1;
  g1[0] = 0x00010000;                                    // data_size=1 (2 bytes)
  g1[1] = (int)((stride_elems & 0xFFFFu) << 16);         // tensor_dim0 lo16
  g1[2] = (int)((stride_elems >> 16) | ((tensor_rows & 0xFFFFu) << 16));
  g1[3] = (int)((tensor_rows >> 16) | (tile_w << 16));   // tensor_dim1 hi | tile_dim0
  g1[4] = (int)(tile_h & 0xFFFFu);                       // tile_dim1 (tile_dim2=0)
  g1[5] = (int)stride_elems;                             // tensor_dim0_stride lo32
  g1[6] = 0;
  g1[7] = 0;
  i32x4 z4 = {0, 0, 0, 0};
  i32x8 z8 = {0, 0, 0, 0, 0, 0, 0, 0};
  __builtin_amdgcn_tensor_load_to_lds(g0, g1, z4, z4, z8, 0);
}
#endif

// ---------------------------------------------------------------------------
// Wo fp32 -> bf16 conversion
__global__ void cvt_wo_kernel(const float* __restrict__ src,
                              unsigned short* __restrict__ dst, int nelem) {
  int i = blockIdx.x * blockDim.x + threadIdx.x;
  if (i < nelem) dst[i] = f2bf(src[i]);
}

// ---------------------------------------------------------------------------
// Per-head QKV projection: out[l,e] = sum_d x[l, h*64+d] * W[e,d]
// mode 0: queries->Qp (row major), 1: keys->Kp (row major), 2: values->VT (transposed)
__global__ void proj_kernel(const float* __restrict__ queries,
                            const float* __restrict__ keys,
                            const float* __restrict__ values,
                            const float* __restrict__ Wq,
                            const float* __restrict__ Wk,
                            const float* __restrict__ Wv,
                            unsigned short* __restrict__ Qp,
                            unsigned short* __restrict__ Kp,
                            unsigned short* __restrict__ VT) {
  int lane = threadIdx.x & 31;
  int wave = threadIdx.x >> 5;
  int mode = blockIdx.z;
  int nh   = blockIdx.y;
  int l0   = blockIdx.x * 128 + wave * 16;
  int n = nh / HEADS, h = nh % HEADS;

  const float* x; const float* W;
  if      (mode == 0) { x = queries; W = Wq; }
  else if (mode == 1) { x = keys;    W = Wk; }
  else                { x = values;  W = Wv; }

  const float* xbase = x + ((size_t)n * LSEQ + l0) * EMBED + h * HDIM;
  bf16x16 a0 = fragA_f32(xbase, EMBED, 0,  lane);
  bf16x16 a1 = fragA_f32(xbase, EMBED, 32, lane);

  int cn = lane & 15, hf = (lane >> 4) & 1;

  #pragma unroll
  for (int t = 0; t < 4; ++t) {
    bf16x16 b0 = fragB_f32(W + (size_t)(t * 16) * HDIM, HDIM, 0,  lane);
    bf16x16 b1 = fragB_f32(W + (size_t)(t * 16) * HDIM, HDIM, 32, lane);
    AccU o;
    #pragma unroll
    for (int r = 0; r < 8; ++r) o.f[r] = 0.0f;
    o.v = wmma_bf(a0, b0, o.v);
    o.v = wmma_bf(a1, b1, o.v);

    int e = t * 16 + cn;            // output column (lane%16 = N in C/D layout)
    if (mode == 2) {
      // transposed store: VT[(nh*64+e)*L + l], rows r -> consecutive l -> b128
      u16x8 pk;
      #pragma unroll
      for (int r = 0; r < 8; ++r) pk[r] = f2bf(o.f[r]);
      *(u16x8*)(VT + ((size_t)nh * HDIM + e) * LSEQ + l0 + 8 * hf) = pk;
    } else {
      unsigned short* dst =
          (mode == 0 ? Qp : Kp) + ((size_t)nh * LSEQ + l0) * HDIM + e;
      #pragma unroll
      for (int r = 0; r < 8; ++r)
        dst[(size_t)(r + 8 * hf) * HDIM] = f2bf(o.f[r]);
    }
  }
}

// ---------------------------------------------------------------------------
// Flash attention: per wave, 16 q-rows x full head dim, streaming 32 keys/iter.
// K/V tiles double-buffered in LDS: the Tensor Data Mover streams tile i+1
// while the 8 waves compute on tile i (wait is s_wait_tensorcnt<=2: in-order
// TDM completion means tile i's two loads have retired).
__global__ void attn_kernel(const unsigned short* __restrict__ Qp,
                            const unsigned short* __restrict__ Kp,
                            const unsigned short* __restrict__ VT,
                            const int* __restrict__ mask,
                            unsigned short* __restrict__ att) {
  __shared__ unsigned short ldsK[2][32 * HDIM];  // 32 keys x 64 d (row major)
  __shared__ unsigned short ldsV[2][HDIM * 32];  // 64 d rows x 32 keys (VT tile)
  __shared__ unsigned short ldsP[8][16 * 32];    // per-wave P repack buffer

  int lane = threadIdx.x & 31;
  int wave = threadIdx.x >> 5;
  int h = blockIdx.y, n = blockIdx.z;
  int nh = n * HEADS + h;
  int q0 = blockIdx.x * 128 + wave * 16;

  const unsigned short* qb = Qp + ((size_t)nh * LSEQ + q0) * HDIM;
  bf16x16 aq0 = fragA_bf16(qb, HDIM, 0,  lane);
  bf16x16 aq1 = fragA_bf16(qb, HDIM, 32, lane);

  int cn = lane & 15, hf = (lane >> 4) & 1;
  const int* mbase = mask + (size_t)n * LSEQ * LSEQ;
  const unsigned short* kg = Kp + (size_t)nh * LSEQ * HDIM;
  const unsigned short* vb = VT + (size_t)nh * HDIM * LSEQ;

  AccU O[4];
  float rm[8], rl[8];
  #pragma unroll
  for (int r = 0; r < 8; ++r) { rm[r] = -3.0e38f; rl[r] = 0.0f; }
  #pragma unroll
  for (int t = 0; t < 4; ++t)
    #pragma unroll
    for (int r = 0; r < 8; ++r) O[t].f[r] = 0.0f;

  const float scale = 0.03125f;  // 1/sqrt(1024)

#ifdef HAVE_TDM
  if (wave == 0) {               // prologue: DMA tile 0 into buffer 0
    tdm_load_2d(ldsK[0], kg, HDIM, 32, HDIM, 32);
    tdm_load_2d(ldsV[0], vb, 32, HDIM, LSEQ, HDIM);
  }
#endif

  for (int k0 = 0; k0 < LSEQ; k0 += 32) {
    int cur = (k0 >> 5) & 1;
    __syncthreads();             // buffer cur^1 no longer being read
#ifdef HAVE_TDM
    if (wave == 0) {
      if (k0 + 32 < LSEQ) {
        tdm_load_2d(ldsK[cur ^ 1], kg + (size_t)(k0 + 32) * HDIM, HDIM, 32, HDIM, 32);
        tdm_load_2d(ldsV[cur ^ 1], vb + k0 + 32, 32, HDIM, LSEQ, HDIM);
        __builtin_amdgcn_s_wait_tensorcnt(2);  // tile k0 complete (in-order)
      } else {
        __builtin_amdgcn_s_wait_tensorcnt(0);
      }
    }
#else
    {
      int t = threadIdx.x;                       // cooperative fallback copy
      *(u16x8*)(ldsK[cur] + t * 8) =
          *(const u16x8*)(kg + (size_t)k0 * HDIM + t * 8);
      int vr = t >> 2, vc = (t & 3) * 8;
      *(u16x8*)(ldsV[cur] + vr * 32 + vc) =
          *(const u16x8*)(vb + (size_t)vr * LSEQ + k0 + vc);
    }
#endif
    __syncthreads();             // tile k0 visible to all waves

    const unsigned short* lk = ldsK[cur];
    const unsigned short* lv = ldsV[cur];

    AccU s0, s1;
    #pragma unroll
    for (int r = 0; r < 8; ++r) { s0.f[r] = 0.0f; s1.f[r] = 0.0f; }
    s0.v = wmma_bf(aq0, fragB_bf16(lk,             HDIM, 0,  lane), s0.v);
    s0.v = wmma_bf(aq1, fragB_bf16(lk,             HDIM, 32, lane), s0.v);
    s1.v = wmma_bf(aq0, fragB_bf16(lk + 16 * HDIM, HDIM, 0,  lane), s1.v);
    s1.v = wmma_bf(aq1, fragB_bf16(lk + 16 * HDIM, HDIM, 32, lane), s1.v);

    float alpha[8];
    #pragma unroll
    for (int r = 0; r < 8; ++r) {
      int qrow = q0 + r + 8 * hf;
      const int* mp = mbase + (size_t)qrow * LSEQ + k0 + cn;
      float e0 = (mp[0]  != 0) ? s0.f[r] * scale : -3.125e18f;
      float e1 = (mp[16] != 0) ? s1.f[r] * scale : -3.125e18f;
      __builtin_prefetch(mp + 32, 0, 0);         // next k-tile mask row
      float mx = fmaxf(e0, e1);
      #pragma unroll
      for (int off = 1; off < 16; off <<= 1)
        mx = fmaxf(mx, __shfl_xor(mx, off, 32));
      float mnew = fmaxf(rm[r], mx);
      float a  = __expf(rm[r] - mnew);
      float p0 = __expf(e0 - mnew);
      float p1 = __expf(e1 - mnew);
      float rs = p0 + p1;
      #pragma unroll
      for (int off = 1; off < 16; off <<= 1)
        rs += __shfl_xor(rs, off, 32);
      rl[r] = rl[r] * a + rs;
      rm[r] = mnew;
      alpha[r] = a;
      int mrow = r + 8 * hf;
      ldsP[wave][mrow * 32 + cn]      = f2bf(p0);
      ldsP[wave][mrow * 32 + 16 + cn] = f2bf(p1);
    }

    #pragma unroll
    for (int t = 0; t < 4; ++t)
      #pragma unroll
      for (int r = 0; r < 8; ++r) O[t].f[r] *= alpha[r];

    bf16x16 pf = fragA_bf16(&ldsP[wave][0], 32, 0, lane);
    #pragma unroll
    for (int t = 0; t < 4; ++t) {
      bf16x16 vf = fragB_bf16(lv + t * 16 * 32, 32, 0, lane);
      O[t].v = wmma_bf(pf, vf, O[t].v);
    }
  }

  float inv[8];
  #pragma unroll
  for (int r = 0; r < 8; ++r) inv[r] = 1.0f / rl[r];

  unsigned short* ob = att + ((size_t)n * LSEQ + q0) * EMBED + h * HDIM;
  #pragma unroll
  for (int t = 0; t < 4; ++t)
    #pragma unroll
    for (int r = 0; r < 8; ++r)
      ob[(size_t)(r + 8 * hf) * EMBED + t * 16 + cn] = f2bf(O[t].f[r] * inv[r]);
}

// ---------------------------------------------------------------------------
// Output projection: out[row,e] = sum_c att[row,c]*Wo[e,c] + bo[e]
// Wo tiles (64 cols x 32 k) double-buffered in LDS via TDM.
__global__ void outproj_kernel(const unsigned short* __restrict__ att,
                               const unsigned short* __restrict__ WoB,
                               const float* __restrict__ bo,
                               float* __restrict__ out) {
  __shared__ unsigned short ldsW[2][HDIM * 32];  // 64 e-rows x 32 k

  int lane = threadIdx.x & 31;
  int wave = threadIdx.x >> 5;
  int row0 = blockIdx.y * 128 + wave * 16;
  int col0 = blockIdx.x * 64;

  const unsigned short* ab = att + (size_t)row0 * EMBED;
  const unsigned short* wg = WoB + (size_t)col0 * EMBED;
  AccU O[4];
  #pragma unroll
  for (int t = 0; t < 4; ++t)
    #pragma unroll
    for (int r = 0; r < 8; ++r) O[t].f[r] = 0.0f;

#ifdef HAVE_TDM
  if (wave == 0) tdm_load_2d(ldsW[0], wg, 32, HDIM, EMBED, HDIM);
#endif

  for (int k0 = 0; k0 < EMBED; k0 += 32) {
    int cur = (k0 >> 5) & 1;
    __syncthreads();
#ifdef HAVE_TDM
    if (wave == 0) {
      if (k0 + 32 < EMBED) {
        tdm_load_2d(ldsW[cur ^ 1], wg + k0 + 32, 32, HDIM, EMBED, HDIM);
        __builtin_amdgcn_s_wait_tensorcnt(1);
      } else {
        __builtin_amdgcn_s_wait_tensorcnt(0);
      }
    }
#else
    {
      int t = threadIdx.x;
      int wr = t >> 2, wc = (t & 3) * 8;
      *(u16x8*)(ldsW[cur] + wr * 32 + wc) =
          *(const u16x8*)(wg + (size_t)wr * EMBED + k0 + wc);
    }
#endif
    __syncthreads();

    bf16x16 af = fragA_bf16(ab, EMBED, k0, lane);
    #pragma unroll
    for (int t = 0; t < 4; ++t) {
      bf16x16 bf_ = fragB_bf16(ldsW[cur] + t * 16 * 32, 32, 0, lane);
      O[t].v = wmma_bf(af, bf_, O[t].v);
    }
  }

  int cn = lane & 15, hf = (lane >> 4) & 1;
  #pragma unroll
  for (int t = 0; t < 4; ++t) {
    float bias = bo[col0 + t * 16 + cn];
    #pragma unroll
    for (int r = 0; r < 8; ++r)
      out[(size_t)(row0 + r + 8 * hf) * EMBED + col0 + t * 16 + cn] =
          O[t].f[r] + bias;
  }
}

// ---------------------------------------------------------------------------
extern "C" void kernel_launch(void* const* d_in, const int* in_sizes, int n_in,
                              void* d_out, int out_size, void* d_ws, size_t ws_size,
                              hipStream_t stream) {
  const float* values  = (const float*)d_in[0];
  const float* keys    = (const float*)d_in[1];
  const float* queries = (const float*)d_in[2];
  const int*   mask    = (const int*)d_in[3];
  const float* Wv = (const float*)d_in[4];
  const float* Wk = (const float*)d_in[5];
  const float* Wq = (const float*)d_in[6];
  const float* Wo = (const float*)d_in[7];
  const float* bo = (const float*)d_in[8];
  float* out = (float*)d_out;

  const size_t perMat = (size_t)NB * HEADS * LSEQ * HDIM;  // 8,388,608 elems
  unsigned short* Qp  = (unsigned short*)d_ws;
  unsigned short* Kp  = Qp + perMat;
  unsigned short* VT  = Kp + perMat;
  unsigned short* att = VT + perMat;
  unsigned short* WoB = att + (size_t)NB * LSEQ * EMBED;

  cvt_wo_kernel<<<(EMBED * EMBED) / 256, 256, 0, stream>>>(Wo, WoB, EMBED * EMBED);
  proj_kernel<<<dim3(LSEQ / 128, NB * HEADS, 3), 256, 0, stream>>>(
      queries, keys, values, Wq, Wk, Wv, Qp, Kp, VT);
  attn_kernel<<<dim3(LSEQ / 128, HEADS, NB), 256, 0, stream>>>(Qp, Kp, VT, mask, att);
  outproj_kernel<<<dim3(EMBED / 64, (NB * LSEQ) / 128), 256, 0, stream>>>(
      att, WoB, bo, out);
}
